// HyperbolicGraphVisitEncoder_82867099009500
// MI455X (gfx1250) — matrix operand order repaired and checked by reading, not compiled, not run on hardware
//
#include <hip/hip_runtime.h>

#define SEQ_L 128
#define DIM 128
#define NSCALES 4
#define BATCH 16                    // rows per async staging batch
#define NBATCH (SEQ_L / BATCH)      // 8
#define RPW (BATCH / 4)             // rows per wave per batch = 4
#define MIN_NORM 1e-15f
#define CLAMP_ABS_EPS 1e-10f
#define ATANH_EPS 1e-7f

typedef __attribute__((ext_vector_type(16))) __bf16 v16bf;
typedef __attribute__((ext_vector_type(8)))  __bf16 v8bf;
typedef __attribute__((ext_vector_type(8)))  float  v8f;

__device__ __forceinline__ unsigned short f2bf(float f) {
  union { float f; unsigned u; } c; c.f = f;
  unsigned r = c.u + 0x7fffu + ((c.u >> 16) & 1u);   // round-to-nearest-even
  return (unsigned short)(r >> 16);
}

__device__ __forceinline__ float artanh_c(float x) {
  x = fminf(fmaxf(x, -1.0f + ATANH_EPS), 1.0f - ATANH_EPS);
  return 0.5f * logf((1.0f + x) / (1.0f - x));
}

// wave32 shuffle reduction + cross-wave combine through 4-slot LDS
__device__ __forceinline__ float block_sum(float v, float* s4, int lane, int wave) {
  #pragma unroll
  for (int off = 16; off > 0; off >>= 1) v += __shfl_xor(v, off, 32);
  __syncthreads();                       // protect s4 against previous use
  if (lane == 0) s4[wave] = v;
  __syncthreads();
  return s4[0] + s4[1] + s4[2] + s4[3];
}

__global__ __launch_bounds__(128) void hyp_midpoint_kernel(
    const int* __restrict__ padded, const float* __restrict__ emb,
    float* __restrict__ out) {
  __shared__ __align__(16) float s_stage[2][BATCH * DIM];      // 2 x 8 KB async staging
  __shared__ __align__(16) unsigned short s_zT[DIM * SEQ_L];   // 32 KB bf16 transposed tile
  __shared__ __align__(32) unsigned short s_coef[SEQ_L];       // lam*mask in bf16
  __shared__ float s_lamm1[SEQ_L];
  __shared__ int   s_idx[SEQ_L];
  __shared__ float s_nom[DIM];
  __shared__ float s_tmp[4];

  const int b    = blockIdx.x;
  const int tid  = threadIdx.x;
  const int lane = tid & 31;
  const int wave = tid >> 5;

  // ---- Phase 1: indices for this batch row ----
  s_idx[tid] = padded[b * SEQ_L + tid];
  __syncthreads();

  // One async B128 per wave moves a full 512B embedding row global->LDS (ASYNCcnt).
  auto issue_batch = [&](int t) {
    const int buf = t & 1;
    #pragma unroll
    for (int r = 0; r < RPW; ++r) {
      const int l   = t * BATCH + wave * RPW + r;
      const int idx = s_idx[l];
      unsigned long long ga =
          (unsigned long long)(size_t)(const void*)(emb + (size_t)idx * DIM + lane * 4);
      unsigned loff =
          (unsigned)(size_t)(const void*)&s_stage[buf][(l & (BATCH - 1)) * DIM + lane * 4];
      asm volatile("global_load_async_to_lds_b128 %0, %1, off"
                   :: "v"(loff), "v"(ga) : "memory");
    }
  };

  // ---- Phase 2: double-buffered async gather + bf16 transpose + conformal factor ----
  issue_batch(0);
  for (int t = 0; t < NBATCH; ++t) {
    if (t + 1 < NBATCH) {
      issue_batch(t + 1);
      // async loads retire in order per wave: <=RPW outstanding == batch t staged
      asm volatile("s_wait_asynccnt %0" :: "i"(RPW) : "memory");
    } else {
      asm volatile("s_wait_asynccnt 0" ::: "memory");
    }
    __syncthreads();                       // batch t visible to all waves
    const int buf = t & 1;
    #pragma unroll
    for (int r = 0; r < RPW; ++r) {
      const int l = t * BATCH + wave * RPW + r;
      const float4 v = *(const float4*)&s_stage[buf][(l & (BATCH - 1)) * DIM + lane * 4];
      float n2p = v.x * v.x + v.y * v.y + v.z * v.z + v.w * v.w;
      #pragma unroll
      for (int off = 16; off > 0; off >>= 1) n2p += __shfl_xor(n2p, off, 32);
      const int d0 = lane * 4;
      s_zT[(d0 + 0) * SEQ_L + l] = f2bf(v.x);
      s_zT[(d0 + 1) * SEQ_L + l] = f2bf(v.y);
      s_zT[(d0 + 2) * SEQ_L + l] = f2bf(v.z);
      s_zT[(d0 + 3) * SEQ_L + l] = f2bf(v.w);
      if (lane == 0) {
        const float lam  = 2.0f / fmaxf(1.0f - n2p, MIN_NORM);
        const bool valid = (s_idx[l] != 0);          // PAD_IDX == 0
        s_coef[l]  = f2bf(valid ? lam : 0.0f);
        s_lamm1[l] = valid ? (lam - 1.0f) : 0.0f;
      }
    }
    __syncthreads();                       // all reads of buf done before re-issue
  }

  // ---- Parallel scalar reductions: denom = sum mask*(lam-1) (1/msum cancels), count ----
  const float den = block_sum(s_lamm1[tid], s_tmp, lane, wave);
  const float cnt = block_sum((s_idx[tid] != 0) ? 1.0f : 0.0f, s_tmp, lane, wave);
  const float sden = (den >= 0.0f ? 1.0f : -1.0f) * fmaxf(fabsf(den), CLAMP_ABS_EPS);

  // ---- Phase 3: nom[d] = sum_l coef_l * z[l][d]  via v_wmma_f32_16x16x32_bf16 ----
  // A[m,k] = zT[16g+m][l0+k]; B[k,n] = coef[l0+k] broadcast over n.
  const int m = lane & 15;
  const int h = lane >> 4;                            // K-half selector (wave32 layout)
  for (int g = 2 * wave; g < 2 * wave + 2; ++g) {     // each wave owns 2 d-groups
    v8f acc = {};
    #pragma unroll
    for (int c = 0; c < 4; ++c) {
      const int l0 = 32 * c;
      const unsigned short* rowp = &s_zT[(16 * g + m) * SEQ_L + l0 + 16 * h];
      v8bf lo = *reinterpret_cast<const v8bf*>(rowp);       // K = 16h + 0..7
      v8bf hi = *reinterpret_cast<const v8bf*>(rowp + 8);   // K = 16h + 8..15
      v16bf a;
      #pragma unroll
      for (int i = 0; i < 8; ++i) { a[i] = lo[i]; a[i + 8] = hi[i]; }
      v16bf bm = *reinterpret_cast<const v16bf*>(&s_coef[l0 + 16 * h]);
      acc = __builtin_amdgcn_wmma_f32_16x16x32_bf16(false, a, false, bm,
                                                    (short)0, acc, false, false);
    }
    if (m == 0) {                                     // all n columns identical
      #pragma unroll
      for (int v = 0; v < 8; ++v)
        s_nom[16 * g + v + 8 * h] = acc[v];
    }
  }
  __syncthreads();

  // ---- Phase 4: two_mean -> mobius 0.5 scalar-mul -> logmap0 (all threads) ----
  const float tm  = s_nom[tid] / sden;                // two_mean[d]
  const float n2  = block_sum(tm * tm, s_tmp, lane, wave);
  const float nrm = fmaxf(sqrtf(n2), MIN_NORM);
  const float tt  = tanhf(0.5f * artanh_c(nrm));      // |mid|
  const float nm  = fmaxf(tt, MIN_NORM);
  const float fac = (tt / nrm) * (artanh_c(nm) / nm);
  const float outv = (cnt == 0.0f) ? 0.0f : tm * fac;
  float* orow = out + (size_t)b * (NSCALES * DIM);
  // weights^s renormalized is identical for every scale s -> replicate.
  #pragma unroll
  for (int s = 0; s < NSCALES; ++s) orow[s * DIM + tid] = outv;
}

extern "C" void kernel_launch(void* const* d_in, const int* in_sizes, int n_in,
                              void* d_out, int out_size, void* d_ws, size_t ws_size,
                              hipStream_t stream) {
  const int*   padded = (const int*)d_in[0];
  const float* emb    = (const float*)d_in[1];
  float*       out    = (float*)d_out;
  const int brows = in_sizes[0] / SEQ_L;              // 8192
  hipLaunchKernelGGL(hyp_midpoint_kernel, dim3(brows), dim3(128), 0, stream,
                     padded, emb, out);
}
